// CNN_GNN_Model_16552803959140
// MI455X (gfx1250) — compile-verified, matrix operand order checked by reading, not run on hardware
//
#include <hip/hip_runtime.h>

typedef __attribute__((ext_vector_type(16))) _Float16 v16h;
typedef __attribute__((ext_vector_type(8)))  _Float16 v8h;
typedef __attribute__((ext_vector_type(8)))  float    v8f;

#define NN 10000
#define NE 160000

// Padded LDS geometry
#define C1_W   42              // conv1 padded row width  (40 + 2 halo)
#define C1_H   44              // conv1 padded rows (40 + halo + 2 overrun rows, zeroed)
#define C1_CH  (C1_W * C1_H)   // 1848 floats per channel
#define C2_W   22              // conv2 padded row width  (20 + 2 halo)
#define C2_H   24              // conv2 padded rows (20 + halo + overrun, zeroed)
#define C2_CH  (C2_W * C2_H)   // 528 floats per channel

// ---------------------------------------------------------------------------
// WMMA implicit-GEMM conv1 (3x3 SAME) + BN + ReLU + maxpool2.
// M (1600 conv pixels) enumerated in pooled-group order: one 2x2 pool cell =
// 4 consecutive accumulator rows of one lane. 16-slot K layout: each ks step
// covers channels {2ks, 2ks+1}; per 8-element group the gathers are
// ds_load_b32 at compile-time offsets from one base address (zero-halo pad).
// Output written directly into conv2's padded layout.
// ---------------------------------------------------------------------------
template<int KS>
__device__ inline void conv1_wmma(const float* __restrict__ in,
                                  const _Float16* __restrict__ wp,
                                  const float* __restrict__ scale,
                                  const float* __restrict__ shift,
                                  float* __restrict__ poolout,
                                  int wave, int lane)
{
    const int h = lane >> 4, l15 = lane & 15;
    for (int mt = wave; mt < 100; mt += 8) {
        const int m = mt * 16 + l15;               // pixel id, pooled-group order
        const int g = m >> 2, s = m & 3;
        const int py = g / 20, px = g % 20;
        const int y = py * 2 + (s >> 1), x = px * 2 + (s & 1);
        const float* base = in + (y + h * 2) * C1_W + x;   // ch0, row y+2h, col x
        v8f acc = {};
#pragma unroll
        for (int ks = 0; ks < KS; ++ks) {
            const float* g0 = base + (ks * 2    ) * C1_CH;
            const float* g1 = base + (ks * 2 + 1) * C1_CH;
            v16h a;
#pragma unroll
            for (int j = 0; j < 4; ++j) {
                a[j]      = (_Float16)g0[j];
                a[4 + j]  = (_Float16)g0[C1_W + j];
                a[8 + j]  = (_Float16)g1[j];
                a[12 + j] = (_Float16)g1[C1_W + j];
            }
            v16h b = *(const v16h*)(wp + (size_t)ks * 512 + lane * 16);
            acc = __builtin_amdgcn_wmma_f32_16x16x32_f16(false, a, false, b,
                                                         (short)0, acc, false, false);
        }
        const int n = l15;
        const float sc = scale[n], sh = shift[n];
        float m0 = fmaxf(fmaxf(acc[0] * sc + sh, acc[1] * sc + sh),
                         fmaxf(acc[2] * sc + sh, acc[3] * sc + sh));
        float m1 = fmaxf(fmaxf(acc[4] * sc + sh, acc[5] * sc + sh),
                         fmaxf(acc[6] * sc + sh, acc[7] * sc + sh));
        m0 = fmaxf(m0, 0.f); m1 = fmaxf(m1, 0.f);
        const int g0i = mt * 4 + h * 2;            // pooled index py*20+px (pair same row)
        const int ppy = g0i / 20, ppx = g0i - ppy * 20;
        float* dst = poolout + n * C2_CH + (ppy + 1) * C2_W + (ppx + 1);
        dst[0] = m0; dst[1] = m1;
    }
}

// ---------------------------------------------------------------------------
// WMMA implicit-GEMM conv2 (16->NOUT, 3x3 SAME on 20x20) + BN + ReLU +
// avgpool4x4: each 16-row M tile is exactly one avgpool cell; in-lane sum of
// 8 accumulator rows + shfl_xor(16) cross-half reduce. K = 16ch*16slots = 256.
// ---------------------------------------------------------------------------
template<int NOUT>
__device__ inline void conv2_wmma(const float* __restrict__ pin,
                                  const _Float16* __restrict__ wp,
                                  const float* __restrict__ scale,
                                  const float* __restrict__ shift,
                                  _Float16* __restrict__ xdst,
                                  int wave, int lane)
{
    const int h = lane >> 4, l15 = lane & 15;
    for (int mt = wave; mt < 25; mt += 8) {        // tile == avgpool cell (oy,ox)
        const int oy = mt / 5, ox = mt % 5;
        const int y = oy * 4 + (l15 >> 2), x = ox * 4 + (l15 & 3);
        const float* base = pin + (y + h * 2) * C2_W + x;
        v8f acc = {};
#pragma unroll
        for (int ks = 0; ks < 8; ++ks) {
            const float* g0 = base + (ks * 2    ) * C2_CH;
            const float* g1 = base + (ks * 2 + 1) * C2_CH;
            v16h a;
#pragma unroll
            for (int j = 0; j < 4; ++j) {
                a[j]      = (_Float16)g0[j];
                a[4 + j]  = (_Float16)g0[C2_W + j];
                a[8 + j]  = (_Float16)g1[j];
                a[12 + j] = (_Float16)g1[C2_W + j];
            }
            v16h b = *(const v16h*)(wp + (size_t)ks * 512 + lane * 16);
            acc = __builtin_amdgcn_wmma_f32_16x16x32_f16(false, a, false, b,
                                                         (short)0, acc, false, false);
        }
        const int n = l15;
        const float sc = (n < NOUT) ? scale[n] : 0.f;
        const float sh = (n < NOUT) ? shift[n] : 0.f;
        float sum = 0.f;
#pragma unroll
        for (int r = 0; r < 8; ++r) sum += fmaxf(acc[r] * sc + sh, 0.f);
        sum += __shfl_xor(sum, 16, 32);            // combine the two M halves
        if (lane < 16 && n < NOUT)
            xdst[n * 25 + mt] = (_Float16)(sum * 0.0625f);
    }
}

// ---------------------------------------------------------------------------
// Fused per-node CNN: both branches, all convs on WMMA, pooling fused.
// LDS: wpk 20480B | par 448B | padded input 44352B | padded pool 33792B
//      = 99072 B total.
// ---------------------------------------------------------------------------
__global__ __launch_bounds__(256) void cnn_kernel(
    const float* __restrict__ veg, const float* __restrict__ cwsi,
    const int* __restrict__ irr, const float* __restrict__ emb,
    const float* __restrict__ vw1, const float* __restrict__ vb1,
    const float* __restrict__ vg1, const float* __restrict__ vbe1,
    const float* __restrict__ vm1, const float* __restrict__ vv1,
    const float* __restrict__ vw2, const float* __restrict__ vb2,
    const float* __restrict__ vg2, const float* __restrict__ vbe2,
    const float* __restrict__ vm2, const float* __restrict__ vv2,
    const float* __restrict__ cw1, const float* __restrict__ cb1,
    const float* __restrict__ cg1, const float* __restrict__ cbe1,
    const float* __restrict__ cm1, const float* __restrict__ cv1,
    const float* __restrict__ cw2, const float* __restrict__ cb2,
    const float* __restrict__ cg2, const float* __restrict__ cbe2,
    const float* __restrict__ cm2, const float* __restrict__ cv2,
    _Float16* __restrict__ xout)
{
    extern __shared__ char smem_raw[];
    _Float16* wpk   = (_Float16*)smem_raw;                     // 10240 halfs
    float*    par   = (float*)(smem_raw + 20480);              // 112 floats
    float*    inb   = (float*)(smem_raw + 20928);              // 11088 floats (6*C1_CH)
    float*    pool1 = (float*)(smem_raw + 20928 + 44352);      // 8448 floats (16*C2_CH)
    const int tid  = threadIdx.x;
    const int node = blockIdx.x;
    const int wave = __builtin_amdgcn_readfirstlane(tid >> 5); // scalar-uniform
    const int lane = tid & 31;

    // ---- zero padded staging buffers (halos + overrun rows + spare chans) ----
    for (int i = tid; i < 6 * C1_CH; i += 256)  inb[i]   = 0.f;
    for (int i = tid; i < 16 * C2_CH; i += 256) pool1[i] = 0.f;
    __syncthreads();

    // ---- pack conv weights (16-slot K layout) + fill veg input + BN params ----
    auto packone = [&](const float* W, _Float16* dst, int CIN, int KS, int Nvalid) {
        for (int idx = tid; idx < KS * 512; idx += 256) {
            int e = idx & 15, ln = (idx >> 4) & 31, ks = idx >> 9;
            int h = ln >> 4, n = ln & 15;
            int k = ks * 32 + (e < 8 ? h * 8 + e : 16 + h * 8 + (e - 8));
            int ci = k >> 4, rem = k & 15, ky = rem >> 2, kx = rem & 3;
            float v = 0.f;
            if (ci < CIN && ky < 3 && kx < 3 && n < Nvalid)
                v = W[((n * CIN + ci) * 3 + ky) * 3 + kx];
            dst[idx] = (_Float16)v;
        }
    };
    packone(vw1, wpk,        5,  3, 16);   // veg conv1  @0     (1536)
    packone(vw2, wpk + 1536, 16, 8, 16);   // veg conv2  @1536  (4096)
    packone(cw1, wpk + 5632, 1,  1, 16);   // cwsi conv1 @5632  (512)
    packone(cw2, wpk + 6144, 16, 8, 8);    // cwsi conv2 @6144  (4096)
    const float* vin = veg + (size_t)node * 8000;
    for (int i = tid; i < 8000; i += 256) {
        int c = i / 1600, rem = i % 1600, yy = rem / 40, xx = rem % 40;
        inb[c * C1_CH + (yy + 1) * C1_W + (xx + 1)] = vin[i];
    }
    if (tid < 16) {                          // folded BN scale/shift (+conv bias)
        float s;
        s = vg1[tid] * rsqrtf(vv1[tid] + 1e-5f);
        par[tid]      = s; par[16 + tid] = vbe1[tid] - vm1[tid]*s + vb1[tid]*s;
        s = vg2[tid] * rsqrtf(vv2[tid] + 1e-5f);
        par[32 + tid] = s; par[48 + tid] = vbe2[tid] - vm2[tid]*s + vb2[tid]*s;
        s = cg1[tid] * rsqrtf(cv1[tid] + 1e-5f);
        par[64 + tid] = s; par[80 + tid] = cbe1[tid] - cm1[tid]*s + cb1[tid]*s;
        if (tid < 8) {
            s = cg2[tid] * rsqrtf(cv2[tid] + 1e-5f);
            par[96 + tid] = s; par[104 + tid] = cbe2[tid] - cm2[tid]*s + cb2[tid]*s;
        }
    }
    __syncthreads();

    _Float16* xrow = xout + (size_t)node * 640;
    // veg conv1 + maxpool -> pool1 (padded interior)
    conv1_wmma<3>(inb, wpk, par, par + 16, pool1, wave, lane);
    __syncthreads();
    // re-zero cwsi staging (2 channels) while pool1 holds veg data
    for (int i = tid; i < 2 * C1_CH; i += 256) inb[i] = 0.f;
    __syncthreads();
    // fill cwsi input + veg conv2 + avgpool -> x[0..399]
    const float* cin = cwsi + (size_t)node * 1600;
    for (int i = tid; i < 1600; i += 256) {
        int yy = i / 40, xx = i % 40;
        inb[(yy + 1) * C1_W + (xx + 1)] = cin[i];
    }
    conv2_wmma<16>(pool1, wpk + 1536, par + 32, par + 48, xrow, wave, lane);
    __syncthreads();
    // cwsi conv1 + maxpool -> pool1
    conv1_wmma<1>(inb, wpk + 5632, par + 64, par + 80, pool1, wave, lane);
    __syncthreads();
    // cwsi conv2 + avgpool -> x[400..599]
    conv2_wmma<8>(pool1, wpk + 6144, par + 96, par + 104, xrow + 400, wave, lane);

    if (tid < 16)              xrow[600 + tid]        = (_Float16)emb[irr[node] * 16 + tid];
    if (tid >= 16 && tid < 40) xrow[616 + (tid - 16)] = (_Float16)0.f;
}

// ---------------------------------------------------------------------------
// SAGE weight pre-pack into WMMA B-fragment order (global, f16, zero k-pad)
// ---------------------------------------------------------------------------
__global__ void pack_w(const float* __restrict__ W, _Float16* __restrict__ Wp,
                       int Nw, int K, int ntiles, int KS)
{
    int idx = blockIdx.x * blockDim.x + threadIdx.x;
    int total = ntiles * KS * 32 * 16;
    if (idx >= total) return;
    int e    = idx & 15;
    int lane = (idx >> 4) & 31;
    int ks   = (idx >> 9) % KS;
    int nt   = idx / (KS * 512);
    int h = lane >> 4;
    int n = nt * 16 + (lane & 15);
    int k = ks * 32 + (e < 8 ? h * 8 + e : 16 + h * 8 + (e - 8));
    float v = (k < K) ? W[(size_t)n * K + k] : 0.f;
    Wp[idx] = (_Float16)v;
}

__global__ void zero_u32(unsigned* __restrict__ p, unsigned long long n)
{
    unsigned long long i = (unsigned long long)blockIdx.x * blockDim.x + threadIdx.x;
    if (i < n) p[i] = 0u;
}

// segment_max via monotone uint-key atomicMax (key 0 == "empty")
__global__ void scatter_max(const _Float16* __restrict__ x,
                            const int* __restrict__ src, const int* __restrict__ dst,
                            unsigned* __restrict__ aggkey, int D, int ldx)
{
    int e = blockIdx.x;
    int s = src[e], d = dst[e];
    for (int f = threadIdx.x; f < D; f += blockDim.x) {
        float v = (float)x[(size_t)s * ldx + f];
        unsigned b = __float_as_uint(v);
        unsigned key = (b & 0x80000000u) ? ~b : (b | 0x80000000u);
        atomicMax(&aggkey[(size_t)d * D + f], key);
    }
}

__global__ void decode_agg(const unsigned* __restrict__ key, _Float16* __restrict__ agg,
                           int D, int Dpad, int total)
{
    int idx = blockIdx.x * blockDim.x + threadIdx.x;
    if (idx >= total) return;
    int n = idx / Dpad, f = idx - n * Dpad;
    float v = 0.f;
    if (f < D) {
        unsigned k = key[(size_t)n * D + f];
        unsigned b = (k & 0x80000000u) ? (k ^ 0x80000000u) : ~k;
        if ((b & 0x7f800000u) != 0x7f800000u) v = __uint_as_float(b);  // !finite -> 0
    }
    agg[idx] = (_Float16)v;
}

// ---------------------------------------------------------------------------
// out = relu(A @ WL^T + X @ WR^T + bias), f16 in/out, f32 WMMA accumulation.
// One wave per 16x16 tile: grid = (M/16, Nout/16).
// ---------------------------------------------------------------------------
__global__ __launch_bounds__(32) void sage_gemm(
    const _Float16* __restrict__ A, const _Float16* __restrict__ X,
    const _Float16* __restrict__ WLp, const _Float16* __restrict__ WRp,
    const float* __restrict__ bias, _Float16* __restrict__ out,
    int Kpad, int KS, int Nout)
{
    const int mt = blockIdx.x, nt = blockIdx.y;
    const int lane = threadIdx.x;
    const int h = lane >> 4, r16 = lane & 15;
    const _Float16* arow = A + (size_t)(mt * 16 + r16) * Kpad;
    const _Float16* xrow = X + (size_t)(mt * 16 + r16) * Kpad;
    const _Float16* wl = WLp + (size_t)nt * KS * 512;
    const _Float16* wr = WRp + (size_t)nt * KS * 512;
    v8f acc = {};
    for (int ks = 0; ks < KS; ++ks) {
        const int koff = ks * 32 + h * 8;
        v8h alo = *(const v8h*)(arow + koff);
        v8h ahi = *(const v8h*)(arow + koff + 16);
        v16h afr;
#pragma unroll
        for (int i = 0; i < 8; ++i) { afr[i] = alo[i]; afr[i + 8] = ahi[i]; }
        v16h bl = *(const v16h*)(wl + ((size_t)ks * 32 + lane) * 16);
        acc = __builtin_amdgcn_wmma_f32_16x16x32_f16(false, afr, false, bl,
                                                     (short)0, acc, false, false);
        v8h xlo = *(const v8h*)(xrow + koff);
        v8h xhi = *(const v8h*)(xrow + koff + 16);
        v16h xfr;
#pragma unroll
        for (int i = 0; i < 8; ++i) { xfr[i] = xlo[i]; xfr[i + 8] = xhi[i]; }
        v16h br = *(const v16h*)(wr + ((size_t)ks * 32 + lane) * 16);
        acc = __builtin_amdgcn_wmma_f32_16x16x32_f16(false, xfr, false, br,
                                                     (short)0, acc, false, false);
    }
    const float bn = bias[nt * 16 + r16];
    const int col = nt * 16 + r16;
#pragma unroll
    for (int r = 0; r < 8; ++r) {
        int row = mt * 16 + r + h * 8;
        float v = fmaxf(acc[r] + bn, 0.f);
        out[(size_t)row * Nout + col] = (_Float16)v;
    }
}

__global__ void fc_out(const _Float16* __restrict__ x2, const float* __restrict__ w,
                       const float* __restrict__ b, float* __restrict__ out, int N)
{
    int i = blockIdx.x * blockDim.x + threadIdx.x;
    if (i >= N) return;
    float acc = b[0];
#pragma unroll
    for (int j = 0; j < 32; ++j) acc += (float)x2[(size_t)i * 32 + j] * w[j];
    out[i] = acc;
}

// ---------------------------------------------------------------------------
extern "C" void kernel_launch(void* const* d_in, const int* in_sizes, int n_in,
                              void* d_out, int out_size, void* d_ws, size_t ws_size,
                              hipStream_t stream)
{
    const float* veg  = (const float*)d_in[0];
    const float* cwsi = (const float*)d_in[1];
    const int*   irr  = (const int*)d_in[2];
    const int*   ei   = (const int*)d_in[3];
    const float* vw1  = (const float*)d_in[4];  const float* vb1 = (const float*)d_in[5];
    const float* vg1  = (const float*)d_in[6];  const float* vbe1= (const float*)d_in[7];
    const float* vm1  = (const float*)d_in[8];  const float* vv1 = (const float*)d_in[9];
    const float* vw2  = (const float*)d_in[10]; const float* vb2 = (const float*)d_in[11];
    const float* vg2  = (const float*)d_in[12]; const float* vbe2= (const float*)d_in[13];
    const float* vm2  = (const float*)d_in[14]; const float* vv2 = (const float*)d_in[15];
    const float* cw1  = (const float*)d_in[16]; const float* cb1 = (const float*)d_in[17];
    const float* cg1  = (const float*)d_in[18]; const float* cbe1= (const float*)d_in[19];
    const float* cm1  = (const float*)d_in[20]; const float* cv1 = (const float*)d_in[21];
    const float* cw2  = (const float*)d_in[22]; const float* cb2 = (const float*)d_in[23];
    const float* cg2  = (const float*)d_in[24]; const float* cbe2= (const float*)d_in[25];
    const float* cm2  = (const float*)d_in[26]; const float* cv2 = (const float*)d_in[27];
    const float* emb  = (const float*)d_in[28];
    const float* s1_wl= (const float*)d_in[29]; const float* s1_bl = (const float*)d_in[30];
    const float* s1_wr= (const float*)d_in[31];
    const float* s2_wl= (const float*)d_in[32]; const float* s2_bl = (const float*)d_in[33];
    const float* s2_wr= (const float*)d_in[34];
    const float* fc_w = (const float*)d_in[35]; const float* fc_b  = (const float*)d_in[36];
    float* out = (float*)d_out;

    char* ws = (char*)d_ws;
    auto carve = [&](size_t bytes) { char* p = ws; ws += (bytes + 255) & ~(size_t)255; return p; };
    _Float16* x    = (_Float16*)carve((size_t)NN * 640 * 2);
    unsigned* ak1  = (unsigned*)carve((size_t)NN * 616 * 4);
    _Float16* ag1  = (_Float16*)carve((size_t)NN * 640 * 2);
    _Float16* x1   = (_Float16*)carve((size_t)NN * 64 * 2);
    unsigned* ak2  = (unsigned*)carve((size_t)NN * 64 * 4);
    _Float16* ag2  = (_Float16*)carve((size_t)NN * 64 * 2);
    _Float16* x2   = (_Float16*)carve((size_t)NN * 32 * 2);
    _Float16* wl1p = (_Float16*)carve((size_t)4 * 20 * 512 * 2);
    _Float16* wr1p = (_Float16*)carve((size_t)4 * 20 * 512 * 2);
    _Float16* wl2p = (_Float16*)carve((size_t)2 * 2 * 512 * 2);
    _Float16* wr2p = (_Float16*)carve((size_t)2 * 2 * 512 * 2);

    pack_w<<<(40960 + 255) / 256, 256, 0, stream>>>(s1_wl, wl1p, 64, 616, 4, 20);
    pack_w<<<(40960 + 255) / 256, 256, 0, stream>>>(s1_wr, wr1p, 64, 616, 4, 20);
    pack_w<<<(2048 + 255) / 256, 256, 0, stream>>>(s2_wl, wl2p, 32, 64, 2, 2);
    pack_w<<<(2048 + 255) / 256, 256, 0, stream>>>(s2_wr, wr2p, 32, 64, 2, 2);

    cnn_kernel<<<NN, 256, 99072, stream>>>(
        veg, cwsi, irr, emb,
        vw1, vb1, vg1, vbe1, vm1, vv1, vw2, vb2, vg2, vbe2, vm2, vv2,
        cw1, cb1, cg1, cbe1, cm1, cv1, cw2, cb2, cg2, cbe2, cm2, cv2, x);

    zero_u32<<<(unsigned)(((size_t)NN * 616 + 255) / 256), 256, 0, stream>>>(ak1, (size_t)NN * 616);
    scatter_max<<<NE, 256, 0, stream>>>(x, ei, ei + NE, ak1, 616, 640);
    decode_agg<<<(NN * 640 + 255) / 256, 256, 0, stream>>>(ak1, ag1, 616, 640, NN * 640);

    dim3 g1(NN / 16, 4);
    sage_gemm<<<g1, 32, 0, stream>>>(ag1, x, wl1p, wr1p, s1_bl, x1, 640, 20, 64);

    zero_u32<<<(unsigned)(((size_t)NN * 64 + 255) / 256), 256, 0, stream>>>(ak2, (size_t)NN * 64);
    scatter_max<<<NE, 256, 0, stream>>>(x1, ei, ei + NE, ak2, 64, 64);
    decode_agg<<<(NN * 64 + 255) / 256, 256, 0, stream>>>(ak2, ag2, 64, 64, NN * 64);

    dim3 g2(NN / 16, 2);
    sage_gemm<<<g2, 32, 0, stream>>>(ag2, x1, wl2p, wr2p, s2_bl, x2, 64, 2, 32);

    fc_out<<<(NN + 255) / 256, 256, 0, stream>>>(x2, fc_w, fc_b, out, NN);
}
